// KAA_40518721470654
// MI455X (gfx1250) — matrix operand order repaired and checked by reading, not compiled, not run on hardware
//
#include <hip/hip_runtime.h>
#include <hip/hip_bf16.h>

typedef __attribute__((ext_vector_type(16))) __bf16 v16bf;
typedef __attribute__((ext_vector_type(8)))  float  v8f;

#define NN 8192
#define KK 32

// ---------------- workspace layout (floats) ----------------
// accum[2] | mc[32] | rsc[32] | mx[8192] | rsx[8192] | E[1024] | M[1024] | q[8192]
// then bf16: XAfrag[8192*32], MXBfrag[8192*32]  (WMMA fragment layout)

__device__ __forceinline__ float softplus_f(float x) {
    return fmaxf(x, 0.0f) + __logf(1.0f + __expf(-fabsf(x)));
}

// -------- init: zero E and accumulators --------
__global__ void k_init(float* E, float* accum) {
    int t = blockIdx.x * blockDim.x + threadIdx.x;
    if (t < 1024) E[t] = 0.0f;
    if (t == 0) { accum[0] = 0.0f; accum[1] = 0.0f; }
}

// -------- C column softmax stats: per column q (32 blocks) --------
__global__ void k_cstats(const float* __restrict__ C, float* mc, float* rsc) {
    __shared__ float red[256];
    const int col = blockIdx.x;
    const int tid = threadIdx.x;
    float m = -3.0e38f;
    for (int j = tid; j < NN; j += 256) m = fmaxf(m, C[(size_t)j * KK + col]);
    red[tid] = m; __syncthreads();
    for (int s = 128; s > 0; s >>= 1) { if (tid < s) red[tid] = fmaxf(red[tid], red[tid + s]); __syncthreads(); }
    const float mcol = red[0]; __syncthreads();
    float ssum = 0.0f;
    for (int j = tid; j < NN; j += 256) ssum += __expf(C[(size_t)j * KK + col] - mcol);
    red[tid] = ssum; __syncthreads();
    for (int s = 128; s > 0; s >>= 1) { if (tid < s) red[tid] += red[tid + s]; __syncthreads(); }
    if (tid == 0) { mc[col] = mcol; rsc[col] = 1.0f / red[0]; }
}

// -------- X per-column (over K=32) softmax stats --------
__global__ void k_xstats(const float* __restrict__ X, float* mx, float* rsx) {
    int j = blockIdx.x * blockDim.x + threadIdx.x;
    float x[KK];
#pragma unroll
    for (int k = 0; k < KK; ++k) x[k] = X[(size_t)k * NN + j];
    float m = x[0];
#pragma unroll
    for (int k = 1; k < KK; ++k) m = fmaxf(m, x[k]);
    float s = 0.0f;
#pragma unroll
    for (int k = 0; k < KK; ++k) s += __expf(x[k] - m);
    mx[j] = m; rsx[j] = 1.0f / s;
}

// -------- E = softmax(X) @ softmax(C)  (32x32), block-partial + atomics --------
__global__ void k_E(const float* __restrict__ X, const float* __restrict__ C,
                    const float* __restrict__ mx, const float* __restrict__ rsx,
                    const float* __restrict__ mc, const float* __restrict__ rsc,
                    float* E) {
    const int p = threadIdx.x >> 5;   // 0..31 (row of Xs)
    const int qc = threadIdx.x & 31;  // 0..31 (col of Cs)
    const float mcq = mc[qc], rscq = rsc[qc];
    const int j0 = blockIdx.x * 256;
    float acc = 0.0f;
    for (int jj = 0; jj < 256; ++jj) {
        int j = j0 + jj;
        float xs = __expf(X[(size_t)p * NN + j] - mx[j]) * rsx[j];
        float cs = __expf(C[(size_t)j * KK + qc] - mcq) * rscq;
        acc += xs * cs;
    }
    atomicAdd(&E[p * KK + qc], acc);
}

// -------- M = E^T E (32x32) --------
__global__ void k_M(const float* __restrict__ E, float* M) {
    __shared__ float Es[1024];
    int t = threadIdx.x;
    Es[t] = E[t];
    __syncthreads();
    int p = t >> 5, qc = t & 31;
    float s = 0.0f;
#pragma unroll
    for (int r = 0; r < KK; ++r) s += Es[r * KK + p] * Es[r * KK + qc];
    M[p * KK + qc] = s;
}

// -------- MX = M@X, q_i = x_i^T M x_i; pack bf16 WMMA fragments --------
__global__ void k_mxq(const float* __restrict__ X, const float* __restrict__ M,
                      float* qv, __bf16* __restrict__ XAfrag, __bf16* __restrict__ MXBfrag) {
    __shared__ float Ms[1024];
    for (int t = threadIdx.x; t < 1024; t += blockDim.x) Ms[t] = M[t];
    __syncthreads();
    const int i = blockIdx.x * blockDim.x + threadIdx.x;   // column index 0..8191
    float x[KK];
#pragma unroll
    for (int k = 0; k < KK; ++k) x[k] = X[(size_t)k * NN + i];
    float mxv[KK];
    float qi = 0.0f;
#pragma unroll
    for (int k = 0; k < KK; ++k) {
        float s = 0.0f;
#pragma unroll
        for (int l = 0; l < KK; ++l) s += Ms[k * KK + l] * x[l];
        mxv[k] = s;
        qi += x[k] * s;
    }
    qv[i] = qi;
    const int tile = i >> 4, m = i & 15;
    const size_t base = (size_t)tile * 512;
    // A-fragment (16-bit A 16x32, ISA 7.12.2): lane m   gets K {0..7,16..23},
    //                                          lane m+16 gets K {8..15,24..31}
#pragma unroll
    for (int e = 0; e < 8; ++e) {
        XAfrag[base + (size_t)m * 16 + e]              = (__bf16)x[e];
        XAfrag[base + (size_t)m * 16 + 8 + e]          = (__bf16)x[16 + e];
        XAfrag[base + (size_t)(m + 16) * 16 + e]       = (__bf16)x[8 + e];
        XAfrag[base + (size_t)(m + 16) * 16 + 8 + e]   = (__bf16)x[24 + e];
    }
    // B-fragment (16-bit B 32x16): lane n gets K {0..7,16..23} of column n; lane n+16 gets K {8..15,24..31}
#pragma unroll
    for (int e = 0; e < 8; ++e) {
        MXBfrag[base + (size_t)m * 16 + e]            = (__bf16)mxv[e];
        MXBfrag[base + (size_t)m * 16 + 8 + e]        = (__bf16)mxv[16 + e];
        MXBfrag[base + (size_t)(m + 16) * 16 + e]     = (__bf16)mxv[8 + e];
        MXBfrag[base + (size_t)(m + 16) * 16 + 8 + e] = (__bf16)mxv[24 + e];
    }
}

// -------- main fused kernel: WMMA G-tile -> theta -> softplus -> reductions --------
#define TJ 8  // j-tiles per wave (A-fragment reuse)
__global__ void __launch_bounds__(256)
k_main(const float* __restrict__ A, const float* __restrict__ beta,
       const float* __restrict__ a_ptr,
       const __bf16* __restrict__ XAfrag, const __bf16* __restrict__ MXBfrag,
       const float* __restrict__ qv, float* accum) {
    const int lane = threadIdx.x & 31;
    const int wave = (blockIdx.x * blockDim.x + threadIdx.x) >> 5;  // 0..32767
    const int itile = wave >> 6;        // 0..511
    const int jgrp  = wave & 63;        // 0..63
    const int i0 = itile << 4;
    const int hi = lane >> 4;
    const int nn = lane & 15;
    const float a_s = a_ptr[0];

    const v16bf afrag = *(const v16bf*)(XAfrag + (size_t)itile * 512 + (size_t)lane * 16);

    float qi[8], bi[8];
#pragma unroll
    for (int v = 0; v < 8; ++v) {
        int irow = i0 + v + 8 * hi;
        qi[v] = qv[irow];
        bi[v] = beta[irow];
    }

    float s1 = 0.0f, s2 = 0.0f;
    for (int jt = 0; jt < TJ; ++jt) {
        const int jtile = jgrp * TJ + jt;
        const int j0 = jtile << 4;
        const v16bf bfrag = *(const v16bf*)(MXBfrag + (size_t)jtile * 512 + (size_t)lane * 16);
        v8f c = {};
        c = __builtin_amdgcn_wmma_f32_16x16x32_bf16(false, afrag, false, bfrag,
                                                    (short)0, c, false, false);
        const int j = j0 + nn;
        const float qj = qv[j];
        const float bj = beta[j];
#pragma unroll
        for (int v = 0; v < 8; ++v) {
            const int irow = i0 + v + 8 * hi;
            const float Av = __builtin_nontemporal_load(&A[(size_t)irow * NN + j]);
            const float theta = bi[v] + bj - a_s * (qi[v] + qj - 2.0f * c[v]);
            s1 += theta * Av;
            s2 += softplus_f(theta);
        }
    }
#pragma unroll
    for (int off = 16; off > 0; off >>= 1) {
        s1 += __shfl_xor(s1, off, 32);
        s2 += __shfl_xor(s2, off, 32);
    }
    if (lane == 0) {
        atomicAdd(&accum[0], s1);
        atomicAdd(&accum[1], s2);
    }
}

// -------- final: add diagonal correction, combine --------
__global__ void k_final(const float* __restrict__ beta, const float* __restrict__ accum,
                        float* out) {
    __shared__ float red[256];
    const int tid = threadIdx.x;
    float s = 0.0f;
    for (int i = tid; i < NN; i += 256) s += softplus_f(2.0f * beta[i]);
    red[tid] = s; __syncthreads();
    for (int st = 128; st > 0; st >>= 1) { if (tid < st) red[tid] += red[tid + st]; __syncthreads(); }
    if (tid == 0) out[0] = 0.5f * (accum[0] - accum[1] + red[0]);
}

extern "C" void kernel_launch(void* const* d_in, const int* in_sizes, int n_in,
                              void* d_out, int out_size, void* d_ws, size_t ws_size,
                              hipStream_t stream) {
    const float* A    = (const float*)d_in[0];
    const float* beta = (const float*)d_in[1];
    const float* a    = (const float*)d_in[2];
    const float* X    = (const float*)d_in[3];
    const float* C    = (const float*)d_in[4];
    float* out = (float*)d_out;

    float* wsf   = (float*)d_ws;
    float* accum = wsf + 0;        // 2
    float* mc    = wsf + 8;        // 32
    float* rsc   = wsf + 40;       // 32
    float* mx    = wsf + 72;       // 8192
    float* rsx   = wsf + 8264;     // 8192
    float* E     = wsf + 16456;    // 1024
    float* M     = wsf + 17480;    // 1024
    float* qv    = wsf + 18504;    // 8192
    __bf16* XAfrag  = (__bf16*)(wsf + 26720);   // 8192*32 bf16, 32B-aligned
    __bf16* MXBfrag = XAfrag + (size_t)NN * KK;

    k_init  <<<5,    256, 0, stream>>>(E, accum);
    k_cstats<<<32,   256, 0, stream>>>(C, mc, rsc);
    k_xstats<<<32,   256, 0, stream>>>(X, mx, rsx);
    k_E     <<<32,  1024, 0, stream>>>(X, C, mx, rsx, mc, rsc, E);
    k_M     <<<1,   1024, 0, stream>>>(E, M);
    k_mxq   <<<32,   256, 0, stream>>>(X, M, qv, XAfrag, MXBfrag);
    k_main  <<<4096, 256, 0, stream>>>(A, beta, a, XAfrag, MXBfrag, qv, accum);
    k_final <<<1,    256, 0, stream>>>(beta, accum, out);
}
